// DMPNNConv_32744830665393
// MI455X (gfx1250) — compile-verified, matrix operand order checked by reading, not run on hardware
//
#include <hip/hip_runtime.h>
#include <stdint.h>

#define N_NODES 62500
#define N_EDGES 1000000
#define LN_EPS 1e-5f

typedef __attribute__((ext_vector_type(16))) __bf16 v16bf;
typedef __attribute__((ext_vector_type(8)))  __bf16 v8bf;
typedef __attribute__((ext_vector_type(8)))  float  v8f;

// Load a 16x32 bf16 WMMA operand fragment from a row-major tile with row
// stride `ld` (elements). Per CDNA5 ISA 16-bit A layout:
//   lanes 0-15  (row = lane):    elems 0-7 = K 0-7,   elems 8-15 = K 16-23
//   lanes 16-31 (row = lane-16): elems 0-7 = K 8-15,  elems 8-15 = K 24-31
__device__ __forceinline__ v16bf load_frag(const __bf16* base, int ld) {
    const int lane = threadIdx.x & 31;
    const int row  = lane & 15;
    const int kh   = (lane >> 4) << 3;          // 0 or 8
    const __bf16* p = base + row * ld + kh;
    v8bf lo = *reinterpret_cast<const v8bf*>(p);
    v8bf hi = *reinterpret_cast<const v8bf*>(p + 16);
    v16bf f;
#pragma unroll
    for (int i = 0; i < 8; ++i) { f[i] = lo[i]; f[i + 8] = hi[i]; }
    return f;
}

__device__ __forceinline__ v8f wmma_bf16(v16bf a, v16bf b, v8f c) {
    return __builtin_amdgcn_wmma_f32_16x16x32_bf16(
        /*neg_a=*/false, a, /*neg_b=*/false, b,
        /*c_mod=*/(short)0, c, /*reuse_a=*/false, /*reuse_b=*/false);
}

// Async global->LDS 16-byte copy (CDNA5 GLOBAL_LOAD_ASYNC_TO_LDS_B128, GV
// addressing: per-lane 64-bit global address, per-lane 32-bit LDS byte addr).
__device__ __forceinline__ void async_copy_b128(unsigned lds_byte_addr,
                                                const void* gptr) {
    asm volatile("global_load_async_to_lds_b128 %0, %1, off"
                 :: "v"(lds_byte_addr),
                    "v"((unsigned long long)(uintptr_t)gptr)
                 : "memory");
}

__device__ __forceinline__ void wait_asynccnt0() {
#if defined(__has_builtin) && __has_builtin(__builtin_amdgcn_s_wait_asynccnt)
    __builtin_amdgcn_s_wait_asynccnt(0);
#else
    asm volatile("s_wait_asynccnt 0x0" ::: "memory");
#endif
}

// ---------------------------------------------------------------------------
// Kernel 0: convert weights to bf16 (W_i padded K 80 -> 96)
// ---------------------------------------------------------------------------
__global__ __launch_bounds__(256) void convert_weights_kernel(
    const float* __restrict__ Wi, const float* __restrict__ Wh,
    const float* __restrict__ Wo,
    __bf16* __restrict__ Wi_b, __bf16* __restrict__ Wh_b, __bf16* __restrict__ Wo_b)
{
    const int tid    = blockIdx.x * 256 + threadIdx.x;
    const int stride = gridDim.x * 256;
    for (int idx = tid; idx < 64 * 96; idx += stride) {
        int r = idx / 96, k = idx - r * 96;
        Wi_b[idx] = (__bf16)(k < 80 ? Wi[r * 80 + k] : 0.f);
    }
    for (int idx = tid; idx < 64 * 64; idx += stride)
        Wh_b[idx] = (__bf16)Wh[idx];
    for (int idx = tid; idx < 64 * 128; idx += stride)
        Wo_b[idx] = (__bf16)Wo[idx];
}

// ---------------------------------------------------------------------------
// Kernel 1: zero the two accumulation buffers (must run every launch)
// ---------------------------------------------------------------------------
__global__ __launch_bounds__(256) void zero_kernel(float* __restrict__ p, long n) {
    long i      = (long)blockIdx.x * blockDim.x + threadIdx.x;
    long stride = (long)gridDim.x * blockDim.x;
    for (; i < n; i += stride) p[i] = 0.f;
}

// ---------------------------------------------------------------------------
// Kernel 2: edge_messages = relu([x[src], e_attr] @ W_i^T); scatter-add to agg
//   block = 128 threads (4 waves), tile = 16 edges, K = 96 (3 k-tiles)
// ---------------------------------------------------------------------------
__global__ __launch_bounds__(128) void edge_message_kernel(
    const float* __restrict__ x, const int* __restrict__ src,
    const int* __restrict__ dst, const float* __restrict__ eattr,
    const __bf16* __restrict__ Wi_b,
    float* __restrict__ edge_msg, float* __restrict__ agg)
{
    __shared__ __align__(16) __bf16 featT[16 * 96];   // [edge][k]  (B^T)
    __shared__ float outT[16 * 64];                   // [edge][m]
    const int tid   = threadIdx.x;
    const int eBase = blockIdx.x * 16;

    for (int idx = tid; idx < 16 * 96; idx += 128) {
        int n = idx / 96, k = idx - n * 96;
        int e = eBase + n;
        float v = 0.f;
        if (k < 64)      v = x[(size_t)src[e] * 64 + k];
        else if (k < 80) v = eattr[(size_t)e * 16 + (k - 64)];
        featT[idx] = (__bf16)v;
    }
    __syncthreads();

    const int wave = tid >> 5, lane = tid & 31;
    v8f acc = {};
#pragma unroll
    for (int kt = 0; kt < 3; ++kt) {
        v16bf a = load_frag(Wi_b + wave * (16 * 96) + kt * 32, 96);
        v16bf b = load_frag(featT + kt * 32, 96);
        acc = wmma_bf16(a, b, acc);
    }
    const int n    = lane & 15;
    const int mrow = (lane >> 4) ? 8 : 0;
#pragma unroll
    for (int i = 0; i < 8; ++i)
        outT[n * 64 + wave * 16 + mrow + i] = fmaxf(acc[i], 0.f);
    __syncthreads();

    for (int idx = tid; idx < 16 * 64; idx += 128) {
        int nn = idx >> 6, m = idx & 63;
        int e  = eBase + nn;
        float v = outT[idx];
        edge_msg[(size_t)e * 64 + m] = v;                     // coalesced
        atomicAdd(&agg[(size_t)dst[e] * 64 + m], v);          // scatter
    }
}

// ---------------------------------------------------------------------------
// Kernel 3: new = LN(relu(agg[src] @ W_h^T) + edge_msg); scatter-add node_msg
//   residual tile staged via async global->LDS DMA (ASYNCcnt path)
// ---------------------------------------------------------------------------
__global__ __launch_bounds__(128) void edge_update_kernel(
    const int* __restrict__ src, const int* __restrict__ dst,
    const float* __restrict__ agg, const __bf16* __restrict__ Wh_b,
    const float* __restrict__ gamma, const float* __restrict__ beta,
    float* __restrict__ new_msg,          // in: edge_msg, out: new_messages
    float* __restrict__ node_msg)
{
    __shared__ __align__(16) __bf16 aggT[16 * 64];    // [edge][k]
    __shared__ __align__(16) float resT[16 * 64];     // residual (edge_msg)
    __shared__ float outT[16 * 64];                   // pre-LN result
    __shared__ float mu[16], rs[16];
    const int tid   = threadIdx.x;
    const int eBase = blockIdx.x * 16;

    // resT <- edge_msg tile: contiguous 4KB, pure copy -> async DMA to LDS
    {
        const unsigned resT_base = (unsigned)(uintptr_t)(void*)resT;
        const char* gsrc = (const char*)(new_msg + (size_t)eBase * 64);
#pragma unroll
        for (int it = 0; it < 2; ++it) {
            unsigned boff = (unsigned)((it * 128 + tid) * 16);
            async_copy_b128(resT_base + boff, gsrc + boff);
        }
    }
    // aggT <- gather agg[src[e]] with f32->bf16 convert (overlaps async DMA)
    for (int idx = tid; idx < 16 * 64; idx += 128) {
        int n = idx >> 6, k = idx & 63;
        int e = eBase + n;
        aggT[idx] = (__bf16)agg[(size_t)src[e] * 64 + k];
    }
    wait_asynccnt0();
    __syncthreads();

    const int wave = tid >> 5, lane = tid & 31;
    v8f acc = {};
#pragma unroll
    for (int kt = 0; kt < 2; ++kt) {
        v16bf a = load_frag(Wh_b + wave * (16 * 64) + kt * 32, 64);
        v16bf b = load_frag(aggT + kt * 32, 64);
        acc = wmma_bf16(a, b, acc);
    }
    const int n    = lane & 15;
    const int mrow = (lane >> 4) ? 8 : 0;
#pragma unroll
    for (int i = 0; i < 8; ++i) {
        int m = wave * 16 + mrow + i;
        outT[n * 64 + m] = fmaxf(acc[i], 0.f) + resT[n * 64 + m];
    }
    __syncthreads();

    if (tid < 16) {                      // per-edge layernorm statistics
        float s = 0.f;
#pragma unroll 8
        for (int m = 0; m < 64; ++m) s += outT[tid * 64 + m];
        float mean = s * (1.f / 64.f);
        float v2 = 0.f;
#pragma unroll 8
        for (int m = 0; m < 64; ++m) {
            float d = outT[tid * 64 + m] - mean;
            v2 += d * d;
        }
        mu[tid] = mean;
        rs[tid] = rsqrtf(v2 * (1.f / 64.f) + LN_EPS);
    }
    __syncthreads();

    for (int idx = tid; idx < 16 * 64; idx += 128) {
        int nn = idx >> 6, m = idx & 63;
        int e  = eBase + nn;
        float v = (outT[idx] - mu[nn]) * rs[nn] * gamma[m] + beta[m];
        new_msg[(size_t)e * 64 + m] = v;
        atomicAdd(&node_msg[(size_t)dst[e] * 64 + m], v);
    }
}

// ---------------------------------------------------------------------------
// Kernel 4: node_output = relu([x, node_msg] @ W_o^T + b_o)
//   tile = 16 nodes, K = 128 (4 k-tiles); bounds-guarded (62500 % 16 != 0)
// ---------------------------------------------------------------------------
__global__ __launch_bounds__(128) void node_output_kernel(
    const float* __restrict__ x, const float* __restrict__ node_msg,
    const __bf16* __restrict__ Wo_b, const float* __restrict__ b_o,
    float* __restrict__ out)
{
    __shared__ __align__(16) __bf16 featT[16 * 128];
    __shared__ float outT[16 * 64];
    const int tid   = threadIdx.x;
    const int nBase = blockIdx.x * 16;

    for (int idx = tid; idx < 16 * 128; idx += 128) {
        int n = idx >> 7, k = idx & 127;
        int node = nBase + n;
        float v = 0.f;
        if (node < N_NODES)
            v = (k < 64) ? x[(size_t)node * 64 + k]
                         : node_msg[(size_t)node * 64 + (k - 64)];
        featT[idx] = (__bf16)v;
    }
    __syncthreads();

    const int wave = tid >> 5, lane = tid & 31;
    v8f acc = {};
#pragma unroll
    for (int kt = 0; kt < 4; ++kt) {
        v16bf a = load_frag(Wo_b + wave * (16 * 128) + kt * 32, 128);
        v16bf b = load_frag(featT + kt * 32, 128);
        acc = wmma_bf16(a, b, acc);
    }
    const int n    = lane & 15;
    const int mrow = (lane >> 4) ? 8 : 0;
#pragma unroll
    for (int i = 0; i < 8; ++i) {
        int m = wave * 16 + mrow + i;
        outT[n * 64 + m] = fmaxf(acc[i] + b_o[m], 0.f);
    }
    __syncthreads();

    for (int idx = tid; idx < 16 * 64; idx += 128) {
        int nn = idx >> 6, m = idx & 63;
        int node = nBase + nn;
        if (node < N_NODES)
            out[(size_t)node * 64 + m] = outT[idx];
    }
}

// ---------------------------------------------------------------------------
extern "C" void kernel_launch(void* const* d_in, const int* in_sizes, int n_in,
                              void* d_out, int out_size, void* d_ws, size_t ws_size,
                              hipStream_t stream) {
    (void)in_sizes; (void)n_in; (void)out_size; (void)ws_size;

    const float* x     = (const float*)d_in[0];
    const int*   eidx  = (const int*)d_in[1];   // [2, E]
    const float* eattr = (const float*)d_in[2];
    const float* Wi    = (const float*)d_in[3]; // [64, 80]
    const float* Wh    = (const float*)d_in[4]; // [64, 64]
    const float* Wo    = (const float*)d_in[5]; // [64, 128]
    const float* b_o   = (const float*)d_in[6];
    const float* gamma = (const float*)d_in[7];
    const float* beta  = (const float*)d_in[8];
    const int* src = eidx;
    const int* dst = eidx + N_EDGES;

    float* node_out = (float*)d_out;                         // [N, 64]
    float* new_msg  = (float*)d_out + (size_t)N_NODES * 64;  // [E, 64]

    // workspace layout (all 16B-aligned):
    float*  agg      = (float*)d_ws;                         // N*64 f32
    float*  node_msg = agg + (size_t)N_NODES * 64;           // N*64 f32
    __bf16* Wi_b     = (__bf16*)(node_msg + (size_t)N_NODES * 64); // 64*96
    __bf16* Wh_b     = Wi_b + 64 * 96;                       // 64*64
    __bf16* Wo_b     = Wh_b + 64 * 64;                       // 64*128

    convert_weights_kernel<<<8, 256, 0, stream>>>(Wi, Wh, Wo, Wi_b, Wh_b, Wo_b);
    zero_kernel<<<2048, 256, 0, stream>>>(agg, (long)N_NODES * 64 * 2);
    edge_message_kernel<<<N_EDGES / 16, 128, 0, stream>>>(
        x, src, dst, eattr, Wi_b, new_msg, agg);
    edge_update_kernel<<<N_EDGES / 16, 128, 0, stream>>>(
        src, dst, agg, Wh_b, gamma, beta, new_msg, node_msg);
    node_output_kernel<<<(N_NODES + 15) / 16, 128, 0, stream>>>(
        x, node_msg, Wo_b, b_o, node_out);
}
